// GaussianImageModel_16518444220960
// MI455X (gfx1250) — matrix operand order repaired and checked by reading, not compiled, not run on hardware
//
#include <hip/hip_runtime.h>
#include <hip/hip_bf16.h>
#include <math.h>

typedef __attribute__((ext_vector_type(16))) _Float16 v16h;
typedef __attribute__((ext_vector_type(8)))  float    v8f;
typedef __attribute__((ext_vector_type(2)))  float    v2f;

#define GH 512
#define GW 512
#define KG 256
#define GEPS 1e-6f
#define LOG2E 1.4426950408889634f

#define WAVES_PER_BLOCK 8
#define PIX_PER_WAVE    16
#define PIX_PER_BLOCK   (WAVES_PER_BLOCK * PIX_PER_WAVE)   // 128
#define NCHUNK          (KG / 32)                          // 8
#define NPAIR           (KG / 2)                           // 128

__global__ __launch_bounds__(256)
void gauss_img_wmma_kernel(const float* __restrict__ grid,
                           const float* __restrict__ mu,
                           const float* __restrict__ log_scales,
                           const float* __restrict__ theta,
                           const float* __restrict__ color_logits,
                           const float* __restrict__ log_amp,
                           float* __restrict__ out)
{
    // Pair-interleaved conic parameters (negated, pre-scaled by log2(e)):
    //   weight = exp2( a*dx^2 + b*dx*dy + c*dy^2 )
    // PP[j] = { mux0,mux1, muy0,muy1, a0,a1, b0,b1, c0,c1, pad,pad }  (48 B, 16B-aligned)
    __shared__ __align__(16) float PP[NPAIR][12];                 // 6 KB
    // B operand pre-packed in WMMA per-lane layout (f16):
    //   Blds[chunk][lane][i], lane = laneHalf*16 + col, K = chunk*32 + laneHalf*16 + i
    __shared__ __align__(32) _Float16 Blds[NCHUNK][32][16];       // 8 KB

    const int t = threadIdx.x;

    // ---------- per-block preamble: one thread per Gaussian ----------
    {
        const int k = t;                       // blockDim.x == 256 == KG
        float c, s;
        __sincosf(theta[k], &s, &c);
        const float sx = __expf(log_scales[2 * k + 0]);
        const float sy = __expf(log_scales[2 * k + 1]);
        const float hx = 0.5f / (sx * sx + GEPS);
        const float hy = 0.5f / (sy * sy + GEPS);
        const float f  = -LOG2E;               // fold negation + exp2 base conversion
        const float an = f * (c * c * hx + s * s * hy);
        const float bn = f * (2.0f * c * s * (hx - hy));
        const float cn = f * (s * s * hx + c * c * hy);

        const int j = k >> 1;
        const int e = k & 1;
        PP[j][0 + e] = mu[2 * k + 0];
        PP[j][2 + e] = mu[2 * k + 1];
        PP[j][4 + e] = an;
        PP[j][6 + e] = bn;
        PP[j][8 + e] = cn;

        // softplus amplitude and sigmoid colors, folded into B operand
        const float la  = log_amp[k];
        const float amp = (la > 0.0f) ? (la + log1pf(__expf(-la))) : log1pf(__expf(la));
        const float bc0 = amp / (1.0f + __expf(-color_logits[3 * k + 0]));
        const float bc1 = amp / (1.0f + __expf(-color_logits[3 * k + 1]));
        const float bc2 = amp / (1.0f + __expf(-color_logits[3 * k + 2]));

        const int chunk    = k >> 5;
        const int laneHalf = (k >> 4) & 1;
        const int i        = k & 15;
        #pragma unroll
        for (int col = 0; col < 16; ++col) {
            float bv;
            if      (col == 0) bv = bc0;
            else if (col == 1) bv = bc1;
            else if (col == 2) bv = bc2;
            else if (col == 3) bv = amp;
            else               bv = 0.0f;
            Blds[chunk][laneHalf * 16 + col][i] = (_Float16)bv;
        }
    }
    __syncthreads();

    // ---------- wave / lane geometry ----------
    const int lane = t & 31;
    const int wave = t >> 5;
    const int tile = blockIdx.x * WAVES_PER_BLOCK + wave;
    const int pix_base = tile * PIX_PER_WAVE;

    const int m  = lane & 15;        // A row / B,D column index
    const int hi = lane >> 4;        // lane half
    const int p  = pix_base + m;
    const float gx = grid[2 * p + 0];
    const float gy = grid[2 * p + 1];
    const v2f gxv = { gx, gx };
    const v2f gyv = { gy, gy };

    // A operand K mapping: lanes 0-15 hold K {0..15}; lanes 16-31 hold K {16..31}
    const int kloA = hi ? 16 : 0;
    const int col  = m;

    v8f acc = {};

    #pragma unroll
    for (int ch = 0; ch < NCHUNK; ++ch) {
        const int kb = ch * 32;

        union { v16h v; _Float16 e[16]; } A;

        // ---- A: 16 Gaussian weights, processed as 8 consecutive-K pairs ----
        #pragma unroll
        for (int i = 0; i < 16; i += 2) {
            const int k = kb + kloA + i;   // even
            const float* pp = &PP[k >> 1][0];
            const v2f muxp = *(const v2f*)(pp + 0);
            const v2f muyp = *(const v2f*)(pp + 2);
            const v2f ap   = *(const v2f*)(pp + 4);
            const v2f bp   = *(const v2f*)(pp + 6);
            const v2f cp   = *(const v2f*)(pp + 8);

            const v2f dx = gxv - muxp;
            const v2f dy = gyv - muyp;
            // q = dx*(a*dx + b*dy) + c*dy*dy   (a,b,c pre-negated & pre-scaled by log2e)
            v2f u = ap * dx;
            u = bp * dy + u;
            u = dx * u;
            const v2f q = cp * (dy * dy) + u;

            A.e[i + 0] = (_Float16)__builtin_amdgcn_exp2f(q.x);
            A.e[i + 1] = (_Float16)__builtin_amdgcn_exp2f(q.y);
        }

        // ---- B: pre-packed per-lane layout, straight from LDS ----
        const v16h Bv = *(const v16h*)&Blds[ch][lane][0];

        acc = __builtin_amdgcn_wmma_f32_16x16x32_f16(
            /*neg_a=*/false, A.v, /*neg_b=*/false, Bv,
            /*c_mod=*/(short)0, acc, /*reuse_a=*/false, /*reuse_b=*/false);
    }

    // ---------- epilogue ----------
    // D layout: lane carries column `col`; rows M = v + hi*8.
    // den (column 3) lives in lane 3 (rows 0-7) and lane 19 (rows 8-15).
    const int den_lane = (hi << 4) | 3;
    #pragma unroll
    for (int v = 0; v < 8; ++v) {
        const float den = __shfl(acc[v], den_lane, 32);
        if (col < 3) {
            float val = acc[v] / (den + GEPS);
            val = fminf(fmaxf(val, 0.0f), 1.0f);
            const int row = (hi << 3) + v;
            out[3 * (pix_base + row) + col] = val;
        }
    }
}

extern "C" void kernel_launch(void* const* d_in, const int* in_sizes, int n_in,
                              void* d_out, int out_size, void* d_ws, size_t ws_size,
                              hipStream_t stream)
{
    const float* grid         = (const float*)d_in[0];  // (H,W,2)
    const float* mu           = (const float*)d_in[1];  // (K,2)
    const float* log_scales   = (const float*)d_in[2];  // (K,2)
    const float* theta        = (const float*)d_in[3];  // (K,)
    const float* color_logits = (const float*)d_in[4];  // (K,3)
    const float* log_amp      = (const float*)d_in[5];  // (K,1)
    float* out                = (float*)d_out;          // (H,W,3)

    const int n_pix  = GH * GW;                         // 262144
    const int blocks = n_pix / PIX_PER_BLOCK;           // 2048

    gauss_img_wmma_kernel<<<blocks, 256, 0, stream>>>(
        grid, mu, log_scales, theta, color_logits, log_amp, out);
}
